// Attention_msa_visual_86517821215101
// MI455X (gfx1250) — matrix operand order repaired and checked by reading, not compiled, not run on hardware
//
#include <hip/hip_runtime.h>
#include <hip/hip_bf16.h>

// MI455X (gfx1250) implementation — all GEMMs as D = A * B^T with bf16 operands
// and f32 WMMA accumulation (v_wmma_f32_16x16x32_bf16), B matrices pre-transposed
// once so every fragment load is a pair of contiguous 128-bit loads.
//
// Live dataflow of the reference:
//   V = x_cls @ W_cls[:,2C:3C]              (NT WMMA GEMM vs Wv^T)
//   per head h: S = Vn_h @ Vn_h^T           (NT WMMA GEMM)
//               attn = softmax(25*S)        (fused; accumulates Ssum/Asum -> bf16 attn)
//               X_h  = attn @ V_h           (NT WMMA GEMM vs V^T -> output cols)
//   out = [concat_h X_h | V]  and  sim_round2(Ssum/8 mask, softmax(Asum/8) renorm)
// q/k/attn_cls/attn_reg and the x_reg/W_reg path are dead in the reference.

typedef __attribute__((ext_vector_type(16))) __bf16 v16bf;
typedef __attribute__((ext_vector_type(8)))  __bf16 v8bf;
typedef __attribute__((ext_vector_type(8)))  float  v8f;

#define NTOK 2048
#define CDIM 1024
#define HDIM 128
#define NH   8

// ---------------- WMMA fragment loaders (ISA 7.12.2 layouts, wave32) --------

// A 16x32 bf16: lane L -> row (L&15); half=L>>4.
// half 0 holds K {0..7, 16..23}; half 1 holds K {8..15, 24..31}.
__device__ __forceinline__ v16bf load_a_frag(const __bf16* __restrict__ A, int lda,
                                             int row0, int k0, int lane) {
  const int r = lane & 15, half = lane >> 4;
  const __bf16* p = A + (size_t)(row0 + r) * lda + k0 + half * 8;
  v8bf lo = *(const v8bf*)p;          // K = k0 + half*8 + [0..7]
  v8bf hi = *(const v8bf*)(p + 16);   // K = k0 + 16 + half*8 + [0..7]
  return __builtin_shufflevector(lo, hi, 0, 1, 2, 3, 4, 5, 6, 7,
                                 8, 9, 10, 11, 12, 13, 14, 15);
}

// B 32x16 bf16 for D = A @ B^T where Bt is (N x K) row-major:
// lane L -> col (L&15); half 0 holds K {0..7, 16..23}, half 1 holds K {8..15, 24..31},
// i.e. two 16-byte runs per lane (two b128 loads).
__device__ __forceinline__ v16bf load_b_frag_nt(const __bf16* __restrict__ Bt, int ldb,
                                                int k0, int col0, int lane) {
  const int c = lane & 15, half = lane >> 4;
  const __bf16* p = Bt + (size_t)(col0 + c) * ldb + k0 + half * 8;
  v8bf lo = *(const v8bf*)p;
  v8bf hi = *(const v8bf*)(p + 16);
  return __builtin_shufflevector(lo, hi, 0, 1, 2, 3, 4, 5, 6, 7,
                                 8, 9, 10, 11, 12, 13, 14, 15);
}

// C/D 16x16 f32: lane L -> col (L&15); VGPR j -> row j + 8*(L>>4).
__device__ __forceinline__ void store_tile(float* __restrict__ D, int ldd,
                                           int row0, int col0, int lane, v8f acc) {
  const int c = lane & 15, half = lane >> 4;
  float* p = D + (size_t)(row0 + half * 8) * ldd + col0 + c;
#pragma unroll
  for (int j = 0; j < 8; ++j) p[(size_t)j * ldd] = acc[j];
}

#define WMMA_BF16(a, b, c) \
  __builtin_amdgcn_wmma_f32_16x16x32_bf16(false, (a), false, (b), (short)0, (c), false, false)

// ---------------- NT GEMM kernel: one wave32 per block, 32x64 tile ----------
// 2 A-fragments x 4 B-fragments -> 8 independent WMMA chains per K-step;
// B fragments are reused across both 16-row strips (better intensity).

__global__ void __launch_bounds__(32)
wmma_gemm_nt(const __bf16* __restrict__ A, int lda,
             const __bf16* __restrict__ Bt, int ldb,
             float* __restrict__ D, int ldd, int K) {
  const int lane = threadIdx.x;
  const int row0 = blockIdx.y * 32;
  const int col0 = blockIdx.x * 64;
  v8f acc00 = {}, acc01 = {}, acc02 = {}, acc03 = {};
  v8f acc10 = {}, acc11 = {}, acc12 = {}, acc13 = {};
  for (int k0 = 0; k0 < K; k0 += 32) {
    v16bf a0 = load_a_frag(A, lda, row0 + 0,  k0, lane);
    v16bf a1 = load_a_frag(A, lda, row0 + 16, k0, lane);
    v16bf b0 = load_b_frag_nt(Bt, ldb, k0, col0 + 0,  lane);
    v16bf b1 = load_b_frag_nt(Bt, ldb, k0, col0 + 16, lane);
    v16bf b2 = load_b_frag_nt(Bt, ldb, k0, col0 + 32, lane);
    v16bf b3 = load_b_frag_nt(Bt, ldb, k0, col0 + 48, lane);
    acc00 = WMMA_BF16(a0, b0, acc00);
    acc01 = WMMA_BF16(a0, b1, acc01);
    acc02 = WMMA_BF16(a0, b2, acc02);
    acc03 = WMMA_BF16(a0, b3, acc03);
    acc10 = WMMA_BF16(a1, b0, acc10);
    acc11 = WMMA_BF16(a1, b1, acc11);
    acc12 = WMMA_BF16(a1, b2, acc12);
    acc13 = WMMA_BF16(a1, b3, acc13);
  }
  store_tile(D, ldd, row0 + 0,  col0 + 0,  lane, acc00);
  store_tile(D, ldd, row0 + 0,  col0 + 16, lane, acc01);
  store_tile(D, ldd, row0 + 0,  col0 + 32, lane, acc02);
  store_tile(D, ldd, row0 + 0,  col0 + 48, lane, acc03);
  store_tile(D, ldd, row0 + 16, col0 + 0,  lane, acc10);
  store_tile(D, ldd, row0 + 16, col0 + 16, lane, acc11);
  store_tile(D, ldd, row0 + 16, col0 + 32, lane, acc12);
  store_tile(D, ldd, row0 + 16, col0 + 48, lane, acc13);
}

// ---------------- block reductions ------------------------------------------

template <int NT>
__device__ __forceinline__ float block_max(float v, float* red, int t) {
  __syncthreads();
  red[t] = v;
  __syncthreads();
#pragma unroll
  for (int o = NT / 2; o > 0; o >>= 1) {
    if (t < o) red[t] = fmaxf(red[t], red[t + o]);
    __syncthreads();
  }
  return red[0];
}

template <int NT>
__device__ __forceinline__ float block_sum(float v, float* red, int t) {
  __syncthreads();
  red[t] = v;
  __syncthreads();
#pragma unroll
  for (int o = NT / 2; o > 0; o >>= 1) {
    if (t < o) red[t] += red[t + o];
    __syncthreads();
  }
  return red[0];
}

// ---------------- prep: bf16 casts (+ Wv transpose) + zero accumulators -----

__global__ void __launch_bounds__(256)
prep_kernel(const float* __restrict__ x, const float* __restrict__ W,
            __bf16* __restrict__ Xbf, __bf16* __restrict__ WvT,
            float* __restrict__ Ssum, float* __restrict__ Asum) {
  const size_t idx = (size_t)blockIdx.x * blockDim.x + threadIdx.x;  // 4M threads
  if (idx < (size_t)NTOK * CDIM) Xbf[idx] = (__bf16)x[idx];
  if (idx < (size_t)CDIM * CDIM) {
    // WvT[c][k] = W[k][2C + c]  (N x K row-major for the NT GEMM)
    const int c = (int)(idx >> 10), k = (int)(idx & (CDIM - 1));
    WvT[idx] = (__bf16)W[(size_t)k * (3 * CDIM) + 2 * CDIM + c];
  }
  Ssum[idx] = 0.0f;
  Asum[idx] = 0.0f;
}

// ---------------- per-(head,row) L2 norm; emit Vn (bf16) + V^T (bf16) -------

__global__ void __launch_bounds__(HDIM)
norm_split_kernel(const float* __restrict__ Vf, __bf16* __restrict__ VbfT,
                  __bf16* __restrict__ Vnbf, float* __restrict__ out) {
  __shared__ float red[HDIM];
  const int h = blockIdx.x, n = blockIdx.y, t = threadIdx.x;
  const size_t i = (size_t)n * CDIM + h * HDIM + t;
  const float v = Vf[i];
  const float ss = block_sum<HDIM>(v * v, red, t);
  const float inv = rsqrtf(ss);
  Vnbf[i] = (__bf16)(v * inv);
  VbfT[(size_t)(h * HDIM + t) * NTOK + n] = (__bf16)v;  // V^T for the attn@V NT GEMM
  out[(size_t)n * (2 * CDIM) + CDIM + h * HDIM + t] = v;  // x_ori half of x_cls_out
}

// ---------------- fused softmax(25*S) + head accumulation -------------------

__global__ void __launch_bounds__(256)
softmax_accum_kernel(const float* __restrict__ S, __bf16* __restrict__ Abf,
                     float* __restrict__ Asum, float* __restrict__ Ssum) {
  __shared__ float red[256];
  const int row = blockIdx.x, t = threadIdx.x;
  const float* srow = S + (size_t)row * NTOK;
  float v[8];
  float m = -3.4e38f;
#pragma unroll
  for (int i = 0; i < 8; ++i) {
    v[i] = srow[t + 256 * i] * 25.0f;  // RAW_SCALE
    m = fmaxf(m, v[i]);
  }
  m = block_max<256>(m, red, t);
  float s = 0.0f;
#pragma unroll
  for (int i = 0; i < 8; ++i) {
    v[i] = __expf(v[i] - m);
    s += v[i];
  }
  s = block_sum<256>(s, red, t);
  const float inv = 1.0f / s;
#pragma unroll
  for (int i = 0; i < 8; ++i) {
    const int j = t + 256 * i;
    const float a = v[i] * inv;
    Abf[(size_t)row * NTOK + j] = (__bf16)a;
    Asum[(size_t)row * NTOK + j] += a;        // heads serialized -> no atomics
    Ssum[(size_t)row * NTOK + j] += srow[j];  // raw attn_cls_raw accumulation
  }
}

// ---------------- sim_round2: mask, softmax, masked renorm ------------------

__global__ void __launch_bounds__(256)
sim_final_kernel(const float* __restrict__ Ssum, const float* __restrict__ Asum,
                 float* __restrict__ out2) {
  __shared__ float red[256];
  const int row = blockIdx.x, t = threadIdx.x;
  const float* arow = Asum + (size_t)row * NTOK;
  const float* srow = Ssum + (size_t)row * NTOK;
  float a[8];
  float m = -3.4e38f;
#pragma unroll
  for (int i = 0; i < 8; ++i) {
    a[i] = arow[t + 256 * i] * 0.125f;  // sim_attn = Asum / H
    m = fmaxf(m, a[i]);
  }
  m = block_max<256>(m, red, t);
  float s = 0.0f;
#pragma unroll
  for (int i = 0; i < 8; ++i) {
    a[i] = __expf(a[i] - m);
    s += a[i];
  }
  s = block_sum<256>(s, red, t);
  const float inv = 1.0f / s;
  float sr[8];
  float ssr = 0.0f;
#pragma unroll
  for (int i = 0; i < 8; ++i) {
    const float mask = (srow[t + 256 * i] * 0.125f > 0.75f) ? 1.0f : 0.0f;
    sr[i] = mask * a[i] * inv;
    ssr += sr[i];
  }
  ssr = block_sum<256>(ssr, red, t);
  const float rinv = 1.0f / ssr;
#pragma unroll
  for (int i = 0; i < 8; ++i)
    out2[(size_t)row * NTOK + t + 256 * i] = sr[i] * rinv;
}

// ---------------- launcher ---------------------------------------------------

extern "C" void kernel_launch(void* const* d_in, const int* in_sizes, int n_in,
                              void* d_out, int out_size, void* d_ws, size_t ws_size,
                              hipStream_t stream) {
  (void)in_sizes; (void)n_in; (void)out_size; (void)ws_size;
  const float* x_cls = (const float*)d_in[0];
  const float* W_cls = (const float*)d_in[2];  // x_reg / W_reg are dead in the reference
  float* out = (float*)d_out;

  char* ws = (char*)d_ws;
  __bf16* Xbf  = (__bf16*)(ws);                        // 2048x1024 bf16  (4 MB)
  __bf16* WvT  = (__bf16*)(ws + (4ull  << 20));        // 1024x1024 bf16  (2 MB) transposed
  float*  Vf   = (float*) (ws + (6ull  << 20));        // 2048x1024 f32   (8 MB)
  __bf16* VbfT = (__bf16*)(ws + (14ull << 20));        // 1024x2048 bf16  (4 MB) transposed
  __bf16* Vnbf = (__bf16*)(ws + (18ull << 20));        // 2048x1024 bf16  (4 MB)
  float*  Sws  = (float*) (ws + (22ull << 20));        // 2048x2048 f32   (16 MB)
  __bf16* Abf  = (__bf16*)(ws + (38ull << 20));        // 2048x2048 bf16  (8 MB)
  float*  Ssum = (float*) (ws + (46ull << 20));        // 2048x2048 f32   (16 MB)
  float*  Asum = (float*) (ws + (62ull << 20));        // 2048x2048 f32   (16 MB)

  // casts + Wv transpose + zero accumulators (ws/d_out are poisoned once)
  prep_kernel<<<(NTOK * NTOK) / 256, 256, 0, stream>>>(x_cls, W_cls, Xbf, WvT, Ssum, Asum);

  // V = x_cls @ W_v  ==  Xbf (2048xK=1024) x WvT (1024xK=1024)^T
  wmma_gemm_nt<<<dim3(CDIM / 64, NTOK / 32), 32, 0, stream>>>(
      Xbf, CDIM, WvT, CDIM, Vf, CDIM, CDIM);

  // per-(row,head) L2 norms; Vn bf16; V^T bf16; x_ori output half
  norm_split_kernel<<<dim3(NH, NTOK), HDIM, 0, stream>>>(Vf, VbfT, Vnbf, out);

  for (int h = 0; h < NH; ++h) {
    // S_h = Vn_h @ Vn_h^T : M=N=2048, K=128
    wmma_gemm_nt<<<dim3(NTOK / 64, NTOK / 32), 32, 0, stream>>>(
        Vnbf + h * HDIM, CDIM, Vnbf + h * HDIM, CDIM, Sws, NTOK, HDIM);
    // attn_h = softmax(25*S_h); accumulate Ssum/Asum; bf16 attn for next GEMM
    softmax_accum_kernel<<<NTOK, 256, 0, stream>>>(Sws, Abf, Asum, Ssum);
    // X_h = attn_h @ V_h  ==  Abf (2048xK=2048) x VbfT_h (128xK=2048)^T
    wmma_gemm_nt<<<dim3(HDIM / 64, NTOK / 32), 32, 0, stream>>>(
        Abf, NTOK, VbfT + (size_t)(h * HDIM) * NTOK, NTOK, out + h * HDIM, 2 * CDIM, NTOK);
  }

  // sim_round2 from head-averaged S (mask) and attn (softmax + masked renorm)
  sim_final_kernel<<<NTOK, 256, 0, stream>>>(Ssum, Asum, out + (size_t)NTOK * NTOK);
}
